// PoseCNN_71004399338037
// MI455X (gfx1250) — compile-verified, matrix operand order checked by reading, not run on hardware
//
#include <hip/hip_runtime.h>
#include <hip/hip_bf16.h>
#include <math.h>

typedef __attribute__((ext_vector_type(16))) __bf16 v16bf;
typedef __attribute__((ext_vector_type(8)))  float  v8f;

__device__ __forceinline__ float sigmoidf_(float x){ return 1.0f/(1.0f+__expf(-x)); }
__device__ __forceinline__ v16bf zero16(){
  v16bf z;
#pragma unroll
  for (int e = 0; e < 16; ++e) z[e] = (__bf16)0.0f;
  return z;
}

// ---------------------------------------------------------------------------
// Weight pre-pack into exact CDNA5 A-fragment order (bf16):
// [cti][ks][lane][e] ; lane l -> row M=l&15 ; e -> K = (l>=16)*8 + (e<8?e:e+8)
// ---------------------------------------------------------------------------
__global__ void pack_conv(const float* __restrict__ w, __bf16* __restrict__ out,
                          int Cout, int Cin, int KH, int nK)
{
  const int KK = KH * KH, Ktot = Cin * KK;
  const int ct = (Cout + 15) >> 4;
  const int total = ct * nK * 512;
  for (int i = blockIdx.x * 256 + threadIdx.x; i < total; i += gridDim.x * 256) {
    int e = i & 15, ln = (i >> 4) & 31, q = i >> 9;
    int ks = q % nK, cti = q / nK;
    int co = cti * 16 + (ln & 15);
    int k  = ks * 32 + ((ln >> 4) << 3) + (e < 8 ? e : e + 8);
    float v = 0.f;
    if (co < Cout && k < Ktot) {
      int ci, ky, kx;
      if (KK == 1) { ci = k; ky = 0; kx = 0; }
      else { ci = k / 9; int rr = k - ci * 9; ky = rr / 3; kx = rr - ky * 3; }
      v = w[((size_t)(co * Cin + ci) * KH + ky) * KH + kx];
    }
    out[i] = (__bf16)v;
  }
}

// deconv weights w(C,O,2,2): pack per (d,k) sub-position: [cti][ks][dk][lane][e]
__global__ void pack_deconv(const float* __restrict__ w, __bf16* __restrict__ out,
                            int C, int O, int nK)
{
  const int ct = (O + 15) >> 4;
  const int total = ct * nK * 4 * 512;
  for (int i = blockIdx.x * 256 + threadIdx.x; i < total; i += gridDim.x * 256) {
    int e = i & 15, ln = (i >> 4) & 31, q = i >> 9;
    int dk = q & 3; q >>= 2;
    int ks = q % nK, cti = q / nK;
    int co = cti * 16 + (ln & 15);
    int k  = ks * 32 + ((ln >> 4) << 3) + (e < 8 ? e : e + 8);
    float v = 0.f;
    if (co < O && k < C) v = w[((size_t)k * O + co) * 4 + dk];
    out[i] = (__bf16)v;
  }
}

__global__ void f32_to_bf16(const float* __restrict__ in, __bf16* __restrict__ out, int n)
{
  int i = blockIdx.x * 256 + threadIdx.x;
  if (i < n) out[i] = (__bf16)in[i];
}

// ---------------------------------------------------------------------------
// Implicit-GEMM conv (KHxKH SAME, stride 1). Block = 8 waves tiling mS x nS
// (mS*nS == 8); each wave owns a 16x32 output tile (two accumulators, two
// back-to-back WMMAs per K-step -> 2x A reuse, half the barriers per WMMA).
// A fragments: coalesced b128 loads from pre-packed weights. B tile (im2col,
// bf16, nF=2*nS fragments) staged in LDS once per K-step, shared by M-waves.
// mode 0: bn(relu(.))->bf16  mode 1: sigmoid->bf16  mode 2: sigmoid(+extra)->f32
// ---------------------------------------------------------------------------
__global__ __launch_bounds__(256) void conv_wmma(
    const __bf16* __restrict__ in, const __bf16* __restrict__ wp,
    const float* __restrict__ bias, const float* __restrict__ bng,
    const float* __restrict__ bnb, const float* __restrict__ extra,
    void* __restrict__ outv,
    int B, int Cin, int Cout, int H, int W, int KH, int wsh, int mode,
    int mS, int nS)
{
  const int pad = KH >> 1, KK = KH * KH;
  const int P = H * W, Ktot = Cin * KK;
  const int nK = (Ktot + 31) >> 5;
  const int nF = nS * 2;                       // B fragments per block
  const int ct = (Cout + 15) >> 4, pt = (P + 15) >> 4;
  const int ctB = (ct + mS - 1) / mS, ptB = (pt + nF - 1) / nF;

  int t = threadIdx.x, lane = t & 31, wv = t >> 5;
  int blk = blockIdx.x;
  int b = blk / (ctB * ptB);
  int r = blk - b * ctB * ptB;
  int cb = r / ptB, pb = r - cb * ptB;
  int mi = wv % mS, ni = wv / mS;
  int cti = cb * mS + mi;
  int pti0 = pb * nF + 2 * ni;                 // this wave's two p-tiles
  int nl = lane & 15, base8 = (lane >> 4) << 3;
  int p_blockbase = pb * nF * 16;
  bool mvalid = cti < ct;
  int p0 = pti0 * 16 + nl, p1 = p0 + 16;
  bool pv0 = (pti0 < pt) && (p0 < P);
  bool pv1 = (pti0 + 1 < pt) && (p1 < P);

  __shared__ alignas(32) __bf16 ldsB[8 * 512];
  v8f acc0 = {0.f,0.f,0.f,0.f,0.f,0.f,0.f,0.f};
  v8f acc1 = {0.f,0.f,0.f,0.f,0.f,0.f,0.f,0.f};

  for (int ks = 0; ks < nK; ++ks) {
    int k0 = ks * 32;
    // cooperative im2col B-tile fill (nF fragments of 512 bf16)
    for (int i = t; i < nF * 512; i += 256) {
      int e = i & 15, ln = (i >> 4) & 31, fr = i >> 9;
      int k = k0 + ((ln >> 4) << 3) + (e < 8 ? e : e + 8);
      int pp = p_blockbase + fr * 16 + (ln & 15);
      __bf16 v = (__bf16)0.0f;
      if (k < Ktot && pp < P) {
        int ci, ky, kx;
        if (KK == 1) { ci = k; ky = 0; kx = 0; }
        else { ci = k / 9; int rr = k - ci * 9; ky = rr / 3; kx = rr - ky * 3; }
        int yy = pp >> wsh, xx = pp & (W - 1);
        int iy = yy + ky - pad, ix = xx + kx - pad;
        if (iy >= 0 && iy < H && ix >= 0 && ix < W)
          v = in[((size_t)(b * Cin + ci)) * P + iy * W + ix];
      }
      ldsB[i] = v;
    }
    __syncthreads();
    v16bf Af = mvalid ? *(const v16bf*)(wp + (((size_t)cti * nK + ks) * 32 + lane) * 16)
                      : zero16();
    v16bf Bf0 = *(const v16bf*)&ldsB[(2 * ni) * 512 + lane * 16];
    v16bf Bf1 = *(const v16bf*)&ldsB[(2 * ni + 1) * 512 + lane * 16];
    acc0 = __builtin_amdgcn_wmma_f32_16x16x32_bf16(false, Af, false, Bf0,
                                                   (short)0, acc0, false, false);
    acc1 = __builtin_amdgcn_wmma_f32_16x16x32_bf16(false, Af, false, Bf1,
                                                   (short)0, acc1, false, false);
    __syncthreads();
  }

  if (!mvalid) return;
  int co_base = cti << 4;
#pragma unroll
  for (int v = 0; v < 8; ++v) {
    int co = co_base + v + base8;
    if (co >= Cout) continue;
    float sconst = (mode == 0) ? (bng[co] * rsqrtf(1.0f + 1e-5f)) : 0.f;
#pragma unroll
    for (int h = 0; h < 2; ++h) {
      bool pv = h ? pv1 : pv0;
      if (!pv) continue;
      int p = h ? p1 : p0;
      float a = (h ? acc1[v] : acc0[v]) + bias[co];
      size_t oidx = ((size_t)b * Cout + co) * P + p;
      if (mode == 0) {
        a = fmaxf(a, 0.f) * sconst + bnb[co];
        ((__bf16*)outv)[oidx] = (__bf16)a;
      } else if (mode == 1) {
        ((__bf16*)outv)[oidx] = (__bf16)sigmoidf_(a);
      } else {
        a += extra[b * Cout + co];
        ((float*)outv)[oidx] = sigmoidf_(a);
      }
    }
  }
}

// ---------------------------------------------------------------------------
// 2x2 deconv: 4 accumulators (one per output sub-position) share each
// LDS-staged B fragment (4 back-to-back WMMAs per K-step); packed A per (d,k).
// ---------------------------------------------------------------------------
__global__ __launch_bounds__(256) void deconv_wmma(
    const __bf16* __restrict__ in, const __bf16* __restrict__ wp,
    const float* __restrict__ bias, __bf16* __restrict__ out,
    int B, int C, int O, int H, int W, int wsh, int mS, int nS)
{
  const int P = H * W;
  const int nK = (C + 31) >> 5;
  const int ct = (O + 15) >> 4, pt = (P + 15) >> 4;
  const int ctB = (ct + mS - 1) / mS, ptB = (pt + nS - 1) / nS;

  int t = threadIdx.x, lane = t & 31, wv = t >> 5;
  int blk = blockIdx.x;
  int b = blk / (ctB * ptB);
  int r = blk - b * ctB * ptB;
  int cb = r / ptB, pb = r - cb * ptB;
  int mi = wv % mS, ni = wv / mS;
  int cti = cb * mS + mi;
  int pti = pb * nS + ni;
  int nl = lane & 15, base8 = (lane >> 4) << 3;
  int p_blockbase = pb * nS * 16;
  bool mvalid = cti < ct;
  int p = pti * 16 + nl;
  bool pv = (pti < pt) && (p < P);
  int y = p >> wsh, x = p & (W - 1);

  __shared__ alignas(32) __bf16 ldsB[8 * 512];
  v8f acc[4];
#pragma unroll
  for (int i = 0; i < 4; ++i) acc[i] = (v8f){0.f,0.f,0.f,0.f,0.f,0.f,0.f,0.f};

  for (int ks = 0; ks < nK; ++ks) {
    int k0 = ks * 32;
    for (int i = t; i < nS * 512; i += 256) {
      int e = i & 15, ln = (i >> 4) & 31, fr = i >> 9;
      int k = k0 + ((ln >> 4) << 3) + (e < 8 ? e : e + 8);
      int pp = p_blockbase + fr * 16 + (ln & 15);
      __bf16 v = (__bf16)0.0f;
      if (k < C && pp < P) v = in[((size_t)b * C + k) * P + pp];
      ldsB[i] = v;
    }
    __syncthreads();
    v16bf Bf = *(const v16bf*)&ldsB[ni * 512 + lane * 16];
#pragma unroll
    for (int dk = 0; dk < 4; ++dk) {
      v16bf Af = mvalid
        ? *(const v16bf*)(wp + ((((size_t)cti * nK + ks) * 4 + dk) * 32 + lane) * 16)
        : zero16();
      acc[dk] = __builtin_amdgcn_wmma_f32_16x16x32_bf16(false, Af, false, Bf,
                                                        (short)0, acc[dk], false, false);
    }
    __syncthreads();
  }

  if (!mvalid || !pv) return;
  int co_base = cti << 4;
  int H2 = H * 2, W2 = W * 2;
#pragma unroll
  for (int dk = 0; dk < 4; ++dk) {
    int d = dk >> 1, k2 = dk & 1;
#pragma unroll
    for (int v = 0; v < 8; ++v) {
      int co = co_base + v + base8;
      if (co >= O) continue;
      float a = fmaxf(acc[dk][v] + bias[co], 0.f);
      out[(((size_t)b * O + co) * H2 + (2 * y + d)) * W2 + (2 * x + k2)] = (__bf16)a;
    }
  }
}

__global__ void maxpool2k(const __bf16* __restrict__ in, __bf16* __restrict__ out,
                          int BC, int H, int W)
{
  int Ho = H >> 1, Wo = W >> 1;
  size_t n = (size_t)BC * Ho * Wo;
  size_t i = (size_t)blockIdx.x * blockDim.x + threadIdx.x;
  if (i >= n) return;
  int xo = (int)(i % Wo); size_t t = i / Wo;
  int yo = (int)(t % Ho); size_t bc = t / Ho;
  const __bf16* pp = in + (bc * H + yo * 2) * (size_t)W + xo * 2;
  float m = fmaxf(fmaxf((float)pp[0], (float)pp[1]),
                  fmaxf((float)pp[W], (float)pp[W + 1]));
  out[i] = (__bf16)m;
}

// soft-argmax over 256x256 heatmaps (temp 0.1) -> (ix, iy) per (b,joint)
__global__ __launch_bounds__(256) void softargmax(const __bf16* __restrict__ hm,
                                                  float* __restrict__ ixy)
{
  const int P = 65536;
  int bj = blockIdx.x, t = threadIdx.x;
  const __bf16* pp = hm + (size_t)bj * P;
  __shared__ float sa[256], sb[256], sc[256];
  float mx = -1e30f;
  for (int i = t; i < P; i += 256) mx = fmaxf(mx, (float)pp[i]);
  sa[t] = mx; __syncthreads();
  for (int s = 128; s > 0; s >>= 1) { if (t < s) sa[t] = fmaxf(sa[t], sa[t + s]); __syncthreads(); }
  float m = sa[0]; __syncthreads();
  float se = 0.f, sx = 0.f, sy = 0.f;
  for (int i = t; i < P; i += 256) {
    float e = __expf(((float)pp[i] - m) * 10.0f);
    se += e; sx += e * (float)(i & 255); sy += e * (float)(i >> 8);
  }
  sa[t] = se; sb[t] = sx; sc[t] = sy; __syncthreads();
  for (int s = 128; s > 0; s >>= 1) {
    if (t < s) { sa[t] += sa[t + s]; sb[t] += sb[t + s]; sc[t] += sc[t + s]; }
    __syncthreads();
  }
  if (t == 0) { ixy[bj * 2] = sb[0] / sa[0]; ixy[bj * 2 + 1] = sc[0] / sa[0]; }
}

__constant__ int NBR_IDX_C[33][4] = {
 {0,0,0,0},{1,0,0,0},{2,0,0,0},{3,0,0,0},{4,0,0,0},{5,0,0,0},{6,0,0,0},{7,0,0,0},
 {8,0,0,0},{9,0,0,0},{10,0,0,0},
 {11,12,23,13},{12,11,24,14},{13,11,15,0},{14,12,16,0},{15,13,0,0},{16,14,0,0},
 {17,0,0,0},{18,0,0,0},{19,0,0,0},{20,0,0,0},{21,0,0,0},{22,0,0,0},
 {23,11,24,25},{24,12,23,26},{25,23,27,0},{26,24,28,0},{27,25,0,0},{28,26,0,0},
 {29,0,0,0},{30,0,0,0},{31,0,0,0},{32,0,0,0}};
__constant__ int NBR_CNT_C[33] =
 {1,1,1,1,1,1,1,1,1,1,1,4,4,3,3,2,2,1,1,1,1,1,1,4,4,3,3,2,2,1,1,1,1};

// bilinear sample + proj + neighbor-mean msg + GRU + per-joint mean bias
__global__ __launch_bounds__(256) void graph_gru(
    const __bf16* __restrict__ fm, const float* __restrict__ ixy,
    const float* __restrict__ proj_w, const float* __restrict__ proj_b,
    const float* __restrict__ msg_w,  const float* __restrict__ msg_b,
    const float* __restrict__ w_ih,   const float* __restrict__ w_hh,
    const float* __restrict__ b_ih,   const float* __restrict__ b_hh,
    float* __restrict__ rbias)
{
  const int J = 33, D = 64, H = 256, W = 256;
  int b = blockIdx.x, t = threadIdx.x;
  __shared__ float joint[J * D], jf[J * D], msgs[J * D];
  for (int idx = t; idx < J * D; idx += 256) {
    int j = idx / D, d = idx - j * D;
    float ix = ixy[(b * J + j) * 2], iy = ixy[(b * J + j) * 2 + 1];
    float x0f = fminf(fmaxf(floorf(ix), 0.f), (float)(W - 1));
    float y0f = fminf(fmaxf(floorf(iy), 0.f), (float)(H - 1));
    float wx = ix - x0f, wy = iy - y0f;
    int x0 = (int)x0f, y0 = (int)y0f;
    int x1 = min(x0 + 1, W - 1), y1 = min(y0 + 1, H - 1);
    const __bf16* ch = fm + ((size_t)b * D + d) * H * W;
    float v00 = (float)ch[y0 * W + x0], v01 = (float)ch[y0 * W + x1];
    float v10 = (float)ch[y1 * W + x0], v11 = (float)ch[y1 * W + x1];
    joint[idx] = v00*(1.f-wx)*(1.f-wy) + v01*wx*(1.f-wy) + v10*(1.f-wx)*wy + v11*wx*wy;
  }
  __syncthreads();
  for (int idx = t; idx < J * D; idx += 256) {
    int j = idx / D, d = idx - j * D;
    float s = proj_b[d];
    for (int k = 0; k < D; ++k) s += joint[j * D + k] * proj_w[d * D + k];
    jf[idx] = s;
  }
  __syncthreads();
  for (int idx = t; idx < J * D; idx += 256) {
    int j = idx / D, d = idx - j * D;
    int c = NBR_CNT_C[j];
    float s = 0.f;
    for (int k = 0; k < c; ++k) s += jf[NBR_IDX_C[j][k] * D + d];
    joint[idx] = s / (float)c;
  }
  __syncthreads();
  for (int idx = t; idx < J * D; idx += 256) {
    int j = idx / D, d = idx - j * D;
    float s = msg_b[d];
    for (int k = 0; k < D; ++k) s += joint[j * D + k] * msg_w[d * D + k];
    msgs[idx] = s;
  }
  __syncthreads();
  for (int idx = t; idx < J * D; idx += 256) {
    int j = idx / D, d = idx - j * D;
    float ir = b_ih[d], iz = b_ih[64 + d], inn = b_ih[128 + d];
    float hr = b_hh[d], hz = b_hh[64 + d], hn  = b_hh[128 + d];
    for (int k = 0; k < D; ++k) {
      float mk = msgs[j * D + k], hk = jf[j * D + k];
      ir  += mk * w_ih[d * 64 + k];
      iz  += mk * w_ih[(64 + d) * 64 + k];
      inn += mk * w_ih[(128 + d) * 64 + k];
      hr  += hk * w_hh[d * 64 + k];
      hz  += hk * w_hh[(64 + d) * 64 + k];
      hn  += hk * w_hh[(128 + d) * 64 + k];
    }
    float rr = sigmoidf_(ir + hr), zz = sigmoidf_(iz + hz);
    float nn = tanhf(inn + rr * hn);
    joint[idx] = (1.f - zz) * nn + zz * jf[idx];
  }
  __syncthreads();
  if (t < J) {
    float s = 0.f;
    for (int d = 0; d < D; ++d) s += joint[t * D + d];
    rbias[b * J + t] = s * (1.f / 64.f);
  }
}

// enc(16,256,64,64) bf16 -> 4x4 avgpool -> 4096 -> 256 relu -> 33 sigmoid
__global__ __launch_bounds__(256) void vis_head(
    const __bf16* __restrict__ enc, const float* __restrict__ w1,
    const float* __restrict__ b1, const float* __restrict__ w2,
    const float* __restrict__ b2, float* __restrict__ out)
{
  int b = blockIdx.x, t = threadIdx.x;
  __shared__ float pooled[4096];
  __shared__ float hv[256];
  for (int idx = t; idx < 4096; idx += 256) {
    int c = idx >> 4, a = (idx >> 2) & 3, q = idx & 3;
    const __bf16* base = enc + (((size_t)b * 256 + c) * 64 + a * 16) * 64 + q * 16;
    float s = 0.f;
    for (int yy = 0; yy < 16; ++yy)
      for (int xx = 0; xx < 16; ++xx) s += (float)base[yy * 64 + xx];
    pooled[idx] = s * (1.f / 256.f);
  }
  __syncthreads();
  {
    float s = b1[t];
    for (int k = 0; k < 4096; ++k) s += pooled[k] * w1[(size_t)t * 4096 + k];
    hv[t] = fmaxf(s, 0.f);
  }
  __syncthreads();
  if (t < 33) {
    float s = b2[t];
    for (int k = 0; k < 256; ++k) s += hv[k] * w2[t * 256 + k];
    out[b * 33 + t] = sigmoidf_(s);
  }
}

static inline int conv_blocks(int B, int Cout, int P, int mS, int nS) {
  int ct = (Cout + 15) / 16, pt = (P + 15) / 16;
  int nF = nS * 2;
  int ctB = (ct + mS - 1) / mS, ptB = (pt + nF - 1) / nF;
  return B * ctB * ptB;
}
static inline int deconv_blocks(int B, int O, int P, int mS, int nS) {
  int ct = (O + 15) / 16, pt = (P + 15) / 16;
  int ctB = (ct + mS - 1) / mS, ptB = (pt + nS - 1) / nS;
  return B * ctB * ptB;
}

extern "C" void kernel_launch(void* const* d_in, const int* in_sizes, int n_in,
                              void* d_out, int out_size, void* d_ws, size_t ws_size,
                              hipStream_t stream) {
  (void)in_sizes; (void)n_in; (void)out_size; (void)ws_size;
  const float* x    = (const float*)d_in[0];
  const float* c1w  = (const float*)d_in[1];  const float* c1b = (const float*)d_in[2];
  const float* bn1g = (const float*)d_in[3];  const float* bn1b= (const float*)d_in[4];
  const float* c2w  = (const float*)d_in[5];  const float* c2b = (const float*)d_in[6];
  const float* bn2g = (const float*)d_in[7];  const float* bn2b= (const float*)d_in[8];
  const float* c3w  = (const float*)d_in[9];  const float* c3b = (const float*)d_in[10];
  const float* bn3g = (const float*)d_in[11]; const float* bn3b= (const float*)d_in[12];
  const float* c4w  = (const float*)d_in[13]; const float* c4b = (const float*)d_in[14];
  const float* bn4g = (const float*)d_in[15]; const float* bn4b= (const float*)d_in[16];
  const float* dc1w = (const float*)d_in[17]; const float* dc1b= (const float*)d_in[18];
  const float* dc2w = (const float*)d_in[19]; const float* dc2b= (const float*)d_in[20];
  const float* hw   = (const float*)d_in[21]; const float* hb  = (const float*)d_in[22];
  const float* pjw  = (const float*)d_in[23]; const float* pjb = (const float*)d_in[24];
  const float* mgw  = (const float*)d_in[25]; const float* mgb = (const float*)d_in[26];
  const float* wih  = (const float*)d_in[27]; const float* whh = (const float*)d_in[28];
  const float* bih  = (const float*)d_in[29]; const float* bhh = (const float*)d_in[30];
  const float* rfw  = (const float*)d_in[31]; const float* rfb = (const float*)d_in[32];
  const float* v1w  = (const float*)d_in[33]; const float* v1b = (const float*)d_in[34];
  const float* v2w  = (const float*)d_in[35]; const float* v2b = (const float*)d_in[36];
  float* out = (float*)d_out;

  // --- workspace bump allocator (256B aligned slots) ---
  char* base = (char*)d_ws;
  auto alloc = [&](size_t bytes) -> char* {
    char* pp = base; base += (bytes + 255) & ~(size_t)255; return pp;
  };
  __bf16* S1  = (__bf16*)alloc((size_t)67108864 * 2); // h2, feat_map
  __bf16* S2  = (__bf16*)alloc((size_t)34603008 * 2); // h1, h3, f, init_hm
  __bf16* S3  = (__bf16*)alloc((size_t)16777216 * 2); // h2p, enc
  __bf16* S4  = (__bf16*)alloc((size_t)8388608  * 2); // h3p
  __bf16* XB  = (__bf16*)alloc((size_t)3145728  * 2); // x in bf16
  __bf16* PW1 = (__bf16*)alloc((size_t)1024   * 2);
  __bf16* PW2 = (__bf16*)alloc((size_t)18432  * 2);
  __bf16* PW3 = (__bf16*)alloc((size_t)73728  * 2);
  __bf16* PW4 = (__bf16*)alloc((size_t)294912 * 2);
  __bf16* PWH = (__bf16*)alloc((size_t)3072   * 2);
  __bf16* PWR = (__bf16*)alloc((size_t)15360  * 2);
  __bf16* PD1 = (__bf16*)alloc((size_t)131072 * 2);
  __bf16* PD2 = (__bf16*)alloc((size_t)32768  * 2);
  float*  IXY = (float*)alloc((size_t)1056 * 4);
  float*  RB  = (float*)alloc((size_t)528 * 4);

  // --- weight packing + input conversion ---
  pack_conv<<<4,   256, 0, stream>>>(c1w, PW1, 32, 3, 3, 1);
  pack_conv<<<72,  256, 0, stream>>>(c2w, PW2, 64, 32, 3, 9);
  pack_conv<<<288, 256, 0, stream>>>(c3w, PW3, 128, 64, 3, 18);
  pack_conv<<<1152,256, 0, stream>>>(c4w, PW4, 256, 128, 3, 36);
  pack_conv<<<12,  256, 0, stream>>>(hw,  PWH, 33, 64, 1, 2);
  pack_conv<<<60,  256, 0, stream>>>(rfw, PWR, 33, 33, 3, 10);
  pack_deconv<<<512,256, 0, stream>>>(dc1w, PD1, 256, 128, 8);
  pack_deconv<<<128,256, 0, stream>>>(dc2w, PD2, 128, 64, 4);
  f32_to_bf16<<<12288, 256, 0, stream>>>(x, XB, 3145728);

  // conv1 + relu + bn1 : XB -> S2   (3->32, 256x256)
  conv_wmma<<<conv_blocks(16,32,65536,2,4),256,0,stream>>>(XB, PW1, c1b, bn1g, bn1b,
      nullptr, S2, 16, 3, 32, 256, 256, 3, 8, 0, 2, 4);
  // conv2 + relu + bn2 : S2 -> S1  (32->64)
  conv_wmma<<<conv_blocks(16,64,65536,4,2),256,0,stream>>>(S2, PW2, c2b, bn2g, bn2b,
      nullptr, S1, 16, 32, 64, 256, 256, 3, 8, 0, 4, 2);
  // maxpool : S1 -> S3
  { size_t n = (size_t)16*64*128*128;
    maxpool2k<<<(int)((n+255)/256),256,0,stream>>>(S1, S3, 16*64, 256, 256); }
  // conv3 + relu + bn3 : S3 -> S2  (64->128, 128x128)
  conv_wmma<<<conv_blocks(16,128,16384,8,1),256,0,stream>>>(S3, PW3, c3b, bn3g, bn3b,
      nullptr, S2, 16, 64, 128, 128, 128, 3, 7, 0, 8, 1);
  // maxpool : S2 -> S4
  { size_t n = (size_t)16*128*64*64;
    maxpool2k<<<(int)((n+255)/256),256,0,stream>>>(S2, S4, 16*128, 128, 128); }
  // conv4 + relu + bn4 : S4 -> S3 (enc)  (128->256, 64x64)
  conv_wmma<<<conv_blocks(16,256,4096,8,1),256,0,stream>>>(S4, PW4, c4b, bn4g, bn4b,
      nullptr, S3, 16, 128, 256, 64, 64, 3, 6, 0, 8, 1);
  // visibility branch: enc -> out tail
  vis_head<<<16,256,0,stream>>>(S3, v1w, v1b, v2w, v2b, out + 34603008);
  // deconv1 + relu : enc(S3) -> S2 (f)   (256->128, 64->128)
  deconv_wmma<<<deconv_blocks(16,128,4096,8,1),256,0,stream>>>(S3, PD1, dc1b, S2,
      16, 256, 128, 64, 64, 6, 8, 1);
  // deconv2 + relu : S2 -> S1 (feat_map) (128->64, 128->256)
  deconv_wmma<<<deconv_blocks(16,64,16384,4,2),256,0,stream>>>(S2, PD2, dc2b, S1,
      16, 128, 64, 128, 128, 7, 4, 2);
  // head 1x1 + sigmoid : S1 -> S2 (init_hm, 33ch)
  conv_wmma<<<conv_blocks(16,33,65536,2,4),256,0,stream>>>(S1, PWH, hb, nullptr, nullptr,
      nullptr, S2, 16, 64, 33, 256, 256, 1, 8, 1, 2, 4);
  // soft-argmax : init_hm -> IXY
  softargmax<<<16*33,256,0,stream>>>(S2, IXY);
  // graph refinement : feat_map(S1) + IXY -> RB
  graph_gru<<<16,256,0,stream>>>(S1, IXY, pjw, pjb, mgw, mgb, wih, whh, bih, bhh, RB);
  // refine conv 3x3 + per-joint bias + sigmoid : init_hm(S2) -> out (fp32)
  conv_wmma<<<conv_blocks(16,33,65536,2,4),256,0,stream>>>(S2, PWR, rfb, nullptr, nullptr,
      RB, out, 16, 33, 33, 256, 256, 3, 8, 2, 2, 4);
}
